// StockLSTM_37417755083191
// MI455X (gfx1250) — matrix operand order repaired and checked by reading, not compiled, not run on hardware
//
#include <hip/hip_runtime.h>
#include <hip/hip_fp16.h>

typedef __attribute__((ext_vector_type(16))) _Float16 v16h;
typedef __attribute__((ext_vector_type(8)))  _Float16 v8h;
typedef __attribute__((ext_vector_type(8)))  float    v8f;

#define TILE_B   32
#define HDIM     64
#define G4       256      // 4*H
#define TSTEPS   256
#define IDIM     5
#define THREADS  256
#define ODIM     25

// ---- LDS layout (all f16) ----
// Wih0pad[256][32], Whh0[256][64], Wih1[256][64], Whh1[256][64],
// X[32][32], H0[2][32][64], H1[2][32][64]   (ping-pong h buffers)
#define HBUF (32 * 64)
#define SMEM_F16_ELEMS (256*32 + 3*256*64 + 32*32 + 4*HBUF)    // 66560
#define SMEM_BYTES (SMEM_F16_ELEMS * 2)                        // 133120

__device__ inline v8f wmma_f16(v16h a, v16h b, v8f c) {
    // v_wmma_f32_16x16x32_f16: D = A(16x32) * B(32x16) + C
    return __builtin_amdgcn_wmma_f32_16x16x32_f16(
        /*neg_a=*/false, a, /*neg_b=*/false, b,
        /*c_mod=*/(short)0, c, /*reuse_a=*/false, /*reuse_b=*/false);
}

// ---- branch-free hardware activations (CDNA5 has v_tanh_f32) ----
__device__ inline float fast_tanh(float x) {
#if __has_builtin(__builtin_amdgcn_tanhf)
    return __builtin_amdgcn_tanhf(x);                 // v_tanh_f32
#else
    // tanh(x) = (1 - e^-2x) / (1 + e^-2x), e^-2x = exp2(-2*log2(e)*x)
    float e = __builtin_amdgcn_exp2f(x * -2.8853900818f);  // v_exp_f32
    return (1.0f - e) * __builtin_amdgcn_rcpf(1.0f + e);   // v_rcp_f32
#endif
}
__device__ inline float fast_sigmoid(float x) {
    // sigmoid(x) = 0.5*tanh(x/2) + 0.5  -> mul + tanh + fma, no branches
    return __builtin_fmaf(0.5f, fast_tanh(0.5f * x), 0.5f);
}

// A fragment: 16x32 (MxK) f16, row-major LDS source [rows][ld].
// lane 0-15:  row=m0+lane,    K = {0..7, 16..23}
// lane 16-31: row=m0+lane-16, K = {8..15, 24..31}
__device__ inline v16h load_A(const _Float16* base, int m0, int ld, int k0) {
    const int l   = threadIdx.x & 31;
    const int row = m0 + (l & 15);
    const int kh  = (l & 16) ? 8 : 0;
    const _Float16* p = base + row * ld + k0 + kh;
    v8h lo = *(const v8h*)(p);          // K = kh .. kh+7
    v8h hi = *(const v8h*)(p + 16);     // K = kh+16 .. kh+23
    v16h r;
#pragma unroll
    for (int i = 0; i < 8; ++i) { r[i] = lo[i]; r[i + 8] = hi[i]; }
    return r;
}

// B fragment: 32x16 (KxN) f16 for gates = h @ W^T, W row-major [256][ld] in LDS.
// B[k][n] = W[n0+n][k]; lane 0-15: n=lane, K = k0+{0..7,16..23}; lane 16-31: K = k0+{8..15,24..31}.
__device__ inline v16h load_B(const _Float16* base, int n0, int ld, int k0) {
    const int l   = threadIdx.x & 31;
    const int row = n0 + (l & 15);
    const int kk  = k0 + ((l & 16) ? 8 : 0);
    const _Float16* p = base + row * ld + kk;
    v8h lo = *(const v8h*)(p);
    v8h hi = *(const v8h*)(p + 16);
    v16h r;
#pragma unroll
    for (int i = 0; i < 8; ++i) { r[i] = lo[i]; r[i + 8] = hi[i]; }
    return r;
}

__global__ __launch_bounds__(THREADS)
void lstm2_fused_kernel(const float* __restrict__ x,
                        const float* __restrict__ Wih0, const float* __restrict__ Whh0,
                        const float* __restrict__ bih0, const float* __restrict__ bhh0,
                        const float* __restrict__ Wih1, const float* __restrict__ Whh1,
                        const float* __restrict__ bih1, const float* __restrict__ bhh1,
                        const float* __restrict__ Wfc,  const float* __restrict__ bfc,
                        float* __restrict__ out)
{
    extern __shared__ char smem[];
    _Float16* sWih0 = (_Float16*)smem;            // [256][32] (K zero-padded from 5)
    _Float16* sWhh0 = sWih0 + 256 * 32;           // [256][64]
    _Float16* sWih1 = sWhh0 + 256 * 64;           // [256][64]
    _Float16* sWhh1 = sWih1 + 256 * 64;           // [256][64]
    _Float16* sX    = sWhh1 + 256 * 64;           // [32][32]
    _Float16* sH0   = sX    + 32 * 32;            // [2][32][64]
    _Float16* sH1   = sH0   + 2 * HBUF;           // [2][32][64]

    const int tid  = threadIdx.x;
    const int wave = tid >> 5;
    const int lane = tid & 31;
    const int b0   = blockIdx.x * TILE_B;

    // ---- one-time preload: weights f32->f16, zero state ----
    for (int i = tid; i < 256 * 32; i += THREADS) {
        int n = i >> 5, k = i & 31;
        sWih0[i] = (k < IDIM) ? (_Float16)Wih0[n * IDIM + k] : (_Float16)0.0f;
    }
    for (int i = tid; i < 256 * 64; i += THREADS) sWhh0[i] = (_Float16)Whh0[i];
    for (int i = tid; i < 256 * 64; i += THREADS) sWih1[i] = (_Float16)Wih1[i];
    for (int i = tid; i < 256 * 64; i += THREADS) sWhh1[i] = (_Float16)Whh1[i];
    for (int i = tid; i < 2 * HBUF; i += THREADS) {
        sH0[i] = (_Float16)0.0f; sH1[i] = (_Float16)0.0f;
    }
    for (int i = tid; i < 32 * 32; i += THREADS) sX[i] = (_Float16)0.0f;

    // ---- per-wave tile ownership: wave = {m-tile, j-tile} ----
    // gate g of columns [j0, j0+16) lives at gate-matrix cols g*64 + j0,
    // so each lane element owns a unique (row, col) across all four gates.
    const int m0    = (wave & 1) * 16;            // batch-row tile
    const int j0    = (wave >> 1) * 16;           // hidden-col tile
    const int col   = j0 + (lane & 15);
    const int rbase = m0 + ((lane & 16) ? 8 : 0); // + v -> C/D row of element v

    float b0r[4], b1r[4];                         // per-lane fused biases
#pragma unroll
    for (int g = 0; g < 4; ++g) {
        b0r[g] = bih0[g * 64 + col] + bhh0[g * 64 + col];
        b1r[g] = bih1[g * 64 + col] + bhh1[g * 64 + col];
    }
    v8f c0 = {};                                  // cell state lives in registers
    v8f c1 = {};
    v8f acc[4];

    __syncthreads();

    for (int t = 0; t < TSTEPS; ++t) {
        const int rp = (t & 1) * HBUF;            // read  h buffer
        const int wp = ((t + 1) & 1) * HBUF;      // write h buffer

        // ---- stage x tile (cols 0..4; rest stay zero) ----
        if (tid < TILE_B * IDIM) {
            int r = tid / IDIM, j = tid - r * IDIM;
            const float* xp = x + ((size_t)(b0 + r) * TSTEPS + (size_t)t) * IDIM + j;
            sX[r * 32 + j] = (_Float16)(*xp);
            if (t + 1 < TSTEPS) __builtin_prefetch(xp + IDIM, 0, 0);  // global_prefetch_b8
        }
        __syncthreads();                          // barrier A: x + prev h1 visible

        // ---- layer 0: gates = Xpad @ Wih0^T + H0 @ Whh0^T ----
        {
            v16h aX  = load_A(sX,       m0, 32, 0);
            v16h a0  = load_A(sH0 + rp, m0, 64, 0);
            v16h a1  = load_A(sH0 + rp, m0, 64, 32);
#pragma unroll
            for (int g = 0; g < 4; ++g) {
                const int n0 = g * 64 + j0;
                v8f a = {};
                a = wmma_f16(aX, load_B(sWih0, n0, 32, 0),  a);
                a = wmma_f16(a0, load_B(sWhh0, n0, 64, 0),  a);
                a = wmma_f16(a1, load_B(sWhh0, n0, 64, 32), a);
                acc[g] = a;
            }
        }
        // ---- layer 0 pointwise in registers -> write h0 into ping-pong buf ----
#pragma unroll
        for (int v = 0; v < 8; ++v) {
            float ig = fast_sigmoid(acc[0][v] + b0r[0]);
            float fg = fast_sigmoid(acc[1][v] + b0r[1]);
            float gg = fast_tanh   (acc[2][v] + b0r[2]);
            float og = fast_sigmoid(acc[3][v] + b0r[3]);
            float c  = fg * c0[v] + ig * gg;
            c0[v] = c;
            sH0[wp + (rbase + v) * 64 + col] = (_Float16)(og * fast_tanh(c));
        }
        __syncthreads();                          // barrier B: new h0 visible

        // ---- layer 1: gates = H0new @ Wih1^T + H1 @ Whh1^T ----
        {
            v16h a0 = load_A(sH0 + wp, m0, 64, 0);
            v16h a1 = load_A(sH0 + wp, m0, 64, 32);
            v16h h0 = load_A(sH1 + rp, m0, 64, 0);
            v16h h1 = load_A(sH1 + rp, m0, 64, 32);
#pragma unroll
            for (int g = 0; g < 4; ++g) {
                const int n0 = g * 64 + j0;
                v8f a = {};
                a = wmma_f16(a0, load_B(sWih1, n0, 64, 0),  a);
                a = wmma_f16(a1, load_B(sWih1, n0, 64, 32), a);
                a = wmma_f16(h0, load_B(sWhh1, n0, 64, 0),  a);
                a = wmma_f16(h1, load_B(sWhh1, n0, 64, 32), a);
                acc[g] = a;
            }
        }
        // ---- layer 1 pointwise -> write h1 into ping-pong buf ----
#pragma unroll
        for (int v = 0; v < 8; ++v) {
            float ig = fast_sigmoid(acc[0][v] + b1r[0]);
            float fg = fast_sigmoid(acc[1][v] + b1r[1]);
            float gg = fast_tanh   (acc[2][v] + b1r[2]);
            float og = fast_sigmoid(acc[3][v] + b1r[3]);
            float c  = fg * c1[v] + ig * gg;
            c1[v] = c;
            sH1[wp + (rbase + v) * 64 + col] = (_Float16)(og * fast_tanh(c));
        }
        // no barrier here: next-iteration barrier A publishes h1 before layer-1 reads it,
        // and ping-pong buffers remove all WAR hazards.
    }
    __syncthreads();

    // ---- final FC: out[b, :] = h1_last @ Wfc^T + bfc ----
    const _Float16* hLast = sH1 + (TSTEPS & 1) * HBUF;   // buffer written at t = TSTEPS-1
    for (int e = tid; e < TILE_B * ODIM; e += THREADS) {
        int r = e / ODIM, o = e - r * ODIM;
        float s = bfc[o];
        const _Float16* hr = hLast + r * HDIM;
        const float* wr = Wfc + o * HDIM;
#pragma unroll 8
        for (int k = 0; k < HDIM; ++k) s += (float)hr[k] * wr[k];
        out[(size_t)(b0 + r) * ODIM + o] = s;
    }
}

extern "C" void kernel_launch(void* const* d_in, const int* in_sizes, int n_in,
                              void* d_out, int out_size, void* d_ws, size_t ws_size,
                              hipStream_t stream) {
    const float* x    = (const float*)d_in[0];
    const float* Wih0 = (const float*)d_in[1];
    const float* Whh0 = (const float*)d_in[2];
    const float* bih0 = (const float*)d_in[3];
    const float* bhh0 = (const float*)d_in[4];
    const float* Wih1 = (const float*)d_in[5];
    const float* Whh1 = (const float*)d_in[6];
    const float* bih1 = (const float*)d_in[7];
    const float* bhh1 = (const float*)d_in[8];
    const float* Wfc  = (const float*)d_in[9];
    const float* bfc  = (const float*)d_in[10];
    float* out = (float*)d_out;

    const int B = in_sizes[0] / (TSTEPS * IDIM);   // 4096
    const int grid = B / TILE_B;                   // 128 workgroups

    // allow >64KB dynamic LDS (~130KB; two WGs still fit in 320KB/WGP)
    (void)hipFuncSetAttribute((const void*)lstm2_fused_kernel,
                              hipFuncAttributeMaxDynamicSharedMemorySize,
                              SMEM_BYTES);

    lstm2_fused_kernel<<<grid, THREADS, SMEM_BYTES, stream>>>(
        x, Wih0, Whh0, bih0, bhh0, Wih1, Whh1, bih1, bhh1, Wfc, bfc, out);
}
